// BidirectionalLandmarkAttention_489626272086
// MI455X (gfx1250) — compile-verified, matrix operand order checked by reading, not run on hardware
//
#include <hip/hip_runtime.h>
#include <math.h>

// ---------------- problem constants ----------------
#define Bc   8
#define Pp   6
#define Nn   4096
#define DIMc 1024
#define Hh   8
#define HDc  128
#define FFc  4096

typedef __attribute__((ext_vector_type(16))) __bf16        bf16x16;
typedef __attribute__((ext_vector_type(8)))  float         f32x8;
typedef __attribute__((ext_vector_type(8)))  unsigned short u16x8;

__device__ __forceinline__ unsigned short f2bf(float f) {
    unsigned int x = __float_as_uint(f);
    unsigned int r = x + 0x7FFFu + ((x >> 16) & 1u);   // round-to-nearest-even
    return (unsigned short)(r >> 16);
}
__device__ __forceinline__ __bf16 bfbits(unsigned short u) {
    return __builtin_bit_cast(__bf16, u);
}

union frag_cast { u16x8 h[2]; bf16x16 v; };

// contiguous 16-bit fragment: two b128 loads
__device__ __forceinline__ bf16x16 load_frag_contig(const unsigned short* p) {
    frag_cast u;
    u.h[0] = *(const u16x8*)(p);
    u.h[1] = *(const u16x8*)(p + 16);
    return u.v;
}

// strided fragment: BSK is a compile-time k-stride so offsets fold into
// immediate offsets of global_load_u16 (max 31*2*BSK bytes, fits 24-bit).
template<int BSK>
__device__ __forceinline__ bf16x16 load_frag(const unsigned short* base, int k0, int khalf) {
    if constexpr (BSK == 1) {
        return load_frag_contig(base + k0 + khalf);
    } else {
        const unsigned short* p = base + (long)(k0 + khalf) * BSK;
        bf16x16 f;
#pragma unroll
        for (int e = 0; e < 16; ++e) {
            int ko = (e < 8) ? e : (e + 8);
            f[e] = bfbits(p[(long)ko * BSK]);
        }
        return f;
    }
}

// ---------------------------------------------------------------------------
// Generic batched WMMA GEMM:  C[z][M,Ncols] = act( A[z] * B[z] + bias )
//   A element (row,k):  Ab + row*lda + k              (k always contiguous)
//   B element (col,k):  Bb + col*bsc + k*BSK          (BSK compile-time: 1 or 1024)
//   C element (row,col):Cb + row*ldc + col
//   z decomposed as (bb,hh) with bb=z/Hdec, hh=z%Hdec; per-operand (SB,SH) strides.
// One wave computes a 32x32 C-tile: 2 A-frags x 2 B-frags -> 4 wmma per K-step.
// Block = 4 waves spanning 128 columns.
// ---------------------------------------------------------------------------
template<int BSK>
__global__ void __launch_bounds__(128)
wmma_gemm_kernel(const unsigned short* __restrict__ A,
                 const unsigned short* __restrict__ Bm,
                 const float* __restrict__ bias,
                 float* __restrict__ Cf,
                 unsigned short* __restrict__ Ch,
                 int M, int Ncols, int Kdim,
                 long lda, long bsc, long ldc,
                 long aSB, long aSH, long bSB, long bSH, long cSB, long cSH,
                 int Hdec, int actGelu, int outBf16)
{
    int z  = blockIdx.z;
    int bb = z / Hdec, hh = z - bb * Hdec;
    const unsigned short* Ab = A  + (long)bb * aSB + (long)hh * aSH;
    const unsigned short* Bb = Bm + (long)bb * bSB + (long)hh * bSH;
    long cbase = (long)bb * cSB + (long)hh * cSH;

    int wave    = threadIdx.x >> 5;
    int lane    = threadIdx.x & 31;
    int tiles32 = (Ncols + 31) >> 5;
    int tileN   = blockIdx.x * 4 + wave;
    if (tileN >= tiles32) return;                // wave-uniform: EXEC stays all-ones

    int row0 = blockIdx.y << 5;
    int col0 = tileN << 5;
    int r    = lane & 15;
    int hi   = lane >> 4;
    int khalf = hi << 3;

    const unsigned short* aRow0 = Ab + (long)(row0 + r) * lda;
    const unsigned short* aRow1 = aRow0 + 16 * lda;
    const unsigned short* bCol0 = Bb + (long)(col0 + r) * bsc;
    const unsigned short* bCol1 = bCol0 + 16 * bsc;

    f32x8 acc00 = {0.f,0.f,0.f,0.f,0.f,0.f,0.f,0.f};
    f32x8 acc01 = acc00, acc10 = acc00, acc11 = acc00;

    for (int k0 = 0; k0 < Kdim; k0 += 32) {
        bf16x16 fa0 = load_frag_contig(aRow0 + k0 + khalf);
        bf16x16 fa1 = load_frag_contig(aRow1 + k0 + khalf);
        bf16x16 fb0 = load_frag<BSK>(bCol0, k0, khalf);
        bf16x16 fb1 = load_frag<BSK>(bCol1, k0, khalf);
        if (k0 + 32 < Kdim) {                    // gfx1250 global_prefetch_b8
            __builtin_prefetch(aRow0 + (k0 + 32), 0, 3);
            __builtin_prefetch(aRow1 + (k0 + 32), 0, 3);
            __builtin_prefetch(bCol0 + (long)(k0 + 32) * BSK, 0, 3);
            __builtin_prefetch(bCol1 + (long)(k0 + 32) * BSK, 0, 3);
        }
        acc00 = __builtin_amdgcn_wmma_f32_16x16x32_bf16(false, fa0, false, fb0, (short)0, acc00, false, false);
        acc01 = __builtin_amdgcn_wmma_f32_16x16x32_bf16(false, fa0, false, fb1, (short)0, acc01, false, false);
        acc10 = __builtin_amdgcn_wmma_f32_16x16x32_bf16(false, fa1, false, fb0, (short)0, acc10, false, false);
        acc11 = __builtin_amdgcn_wmma_f32_16x16x32_bf16(false, fa1, false, fb1, (short)0, acc11, false, false);
    }

#pragma unroll
    for (int i = 0; i < 2; ++i) {
#pragma unroll
        for (int j = 0; j < 2; ++j) {
            const f32x8& acc = (i == 0) ? (j == 0 ? acc00 : acc01)
                                        : (j == 0 ? acc10 : acc11);
            int cc = col0 + j * 16 + r;
#pragma unroll
            for (int v = 0; v < 8; ++v) {
                int rr = row0 + i * 16 + v + (hi << 3);
                if (rr < M && cc < Ncols) {
                    float val = acc[v];
                    if (bias) val += bias[cc];
                    if (actGelu) val = 0.5f * val * (1.0f + erff(val * 0.70710678118654752f));
                    long idx = cbase + (long)rr * ldc + cc;
                    if (outBf16) Ch[idx] = f2bf(val);
                    else         Cf[idx] = val;
                }
            }
        }
    }
}

// -------------------- fp32 -> bf16 elementwise --------------------
__global__ void f2bf_kernel(const float* __restrict__ x, unsigned short* __restrict__ y, long n)
{
    long i = (long)blockIdx.x * blockDim.x + threadIdx.x;
    if (i < n) y[i] = f2bf(x[i]);
}

// ---------------- weight convert+transpose: Wt[n*K+k] = bf16(W[k*N+n]) ----------------
__global__ void wtrans_kernel(const float* __restrict__ W, unsigned short* __restrict__ Wt,
                              int K, int Nout)
{
    int k = blockIdx.x * 256 + threadIdx.x;
    int n = blockIdx.y;
    if (k < K) Wt[(long)n * K + k] = f2bf(W[(long)k * Nout + n]);
}

// ---------------- residual-add LayerNorm (row length = 1024) ----------------
__global__ void __launch_bounds__(256)
ln_kernel(const float* __restrict__ x, const float* __restrict__ delta,
          const float* __restrict__ g, const float* __restrict__ bta,
          float* __restrict__ y32, unsigned short* __restrict__ y16)
{
    __shared__ float red[256];
    int row = blockIdx.x;
    int t   = threadIdx.x;
    long base = (long)row * DIMc;
    float v[4]; float s = 0.f;
#pragma unroll
    for (int i = 0; i < 4; ++i) {
        int idx = t + i * 256;
        v[i] = x[base + idx] + delta[base + idx];
        s += v[i];
    }
    red[t] = s; __syncthreads();
    for (int o = 128; o > 0; o >>= 1) { if (t < o) red[t] += red[t + o]; __syncthreads(); }
    float mean = red[0] * (1.0f / DIMc);
    __syncthreads();
    float vs = 0.f;
#pragma unroll
    for (int i = 0; i < 4; ++i) { float d = v[i] - mean; vs += d * d; }
    red[t] = vs; __syncthreads();
    for (int o = 128; o > 0; o >>= 1) { if (t < o) red[t] += red[t + o]; __syncthreads(); }
    float rstd = rsqrtf(red[0] * (1.0f / DIMc) + 1e-5f);
    __syncthreads();
#pragma unroll
    for (int i = 0; i < 4; ++i) {
        int idx = t + i * 256;
        float yv = (v[i] - mean) * rstd * g[idx] + bta[idx];
        y32[base + idx] = yv;
        if (y16) y16[base + idx] = f2bf(yv);
    }
}

// ---------------- softmax over 4096 keys (p2i); applies HD^-0.5; writes bf16 probs ----------------
__global__ void __launch_bounds__(256)
softmax_p2i_kernel(const float* __restrict__ scores, unsigned short* __restrict__ probs)
{
    __shared__ float red[256];
    const float alpha = 0.08838834764831845f;   // 1/sqrt(128)
    long base = (long)blockIdx.x * Nn;          // rows [B,H,P] contiguous
    int t = threadIdx.x;
    float v[16]; float m = -3.4e38f;
#pragma unroll
    for (int i = 0; i < 16; ++i) { v[i] = alpha * scores[base + t + i * 256]; m = fmaxf(m, v[i]); }
    red[t] = m; __syncthreads();
    for (int o = 128; o > 0; o >>= 1) { if (t < o) red[t] = fmaxf(red[t], red[t + o]); __syncthreads(); }
    m = red[0]; __syncthreads();
    float s = 0.f;
#pragma unroll
    for (int i = 0; i < 16; ++i) { v[i] = expf(v[i] - m); s += v[i]; }
    red[t] = s; __syncthreads();
    for (int o = 128; o > 0; o >>= 1) { if (t < o) red[t] += red[t + o]; __syncthreads(); }
    float inv = 1.0f / red[0];
#pragma unroll
    for (int i = 0; i < 16; ++i) probs[base + t + i * 256] = f2bf(v[i] * inv);
}

// ---------------- softmax over 6 keys (i2p); pads probs row to 32 with zeros ----------------
__global__ void softmax_i2p_kernel(const float* __restrict__ scores, unsigned short* __restrict__ probs)
{
    long r = (long)blockIdx.x * blockDim.x + threadIdx.x;   // (b,h,n) row
    if (r >= (long)Bc * Hh * Nn) return;
    const float alpha = 0.08838834764831845f;
    const float* s = scores + r * 6;
    unsigned short* p = probs + r * 32;
    float v[6]; float m = -3.4e38f;
#pragma unroll
    for (int k = 0; k < 6; ++k) { v[k] = alpha * s[k]; m = fmaxf(m, v[k]); }
    float sum = 0.f;
#pragma unroll
    for (int k = 0; k < 6; ++k) { v[k] = expf(v[k] - m); sum += v[k]; }
    float inv = 1.0f / sum;
#pragma unroll
    for (int k = 0; k < 6; ++k) p[k] = f2bf(v[k] * inv);
#pragma unroll
    for (int k = 6; k < 32; ++k) p[k] = 0;
}

// ---------------- tiny p2p self-attention with relative-position bias ----------------
// grid = B*H*P blocks, block = 128 (= HD). Q/K/V fp32 in [B,P,H,HD] layout.
__global__ void __launch_bounds__(128)
p2p_attn_kernel(const float* __restrict__ Q, const float* __restrict__ K,
                const float* __restrict__ V, const float* __restrict__ rel,
                unsigned short* __restrict__ outBf)
{
    __shared__ float red[128];
    __shared__ float sc[Pp];
    int id = blockIdx.x;
    int q = id % Pp; int h = (id / Pp) % Hh; int b = id / (Pp * Hh);
    int d = threadIdx.x;
    const float alpha = 0.08838834764831845f;
    float qd = Q[(long)(b * Pp + q) * DIMc + h * HDc + d] * alpha;
    for (int k = 0; k < Pp; ++k) {
        float t = qd * (K[(long)(b * Pp + k) * DIMc + h * HDc + d] +
                        rel[((long)q * Pp + k) * HDc + d]);
        red[d] = t; __syncthreads();
        for (int o = 64; o > 0; o >>= 1) { if (d < o) red[d] += red[d + o]; __syncthreads(); }
        if (d == 0) sc[k] = red[0];
        __syncthreads();
    }
    float m = sc[0];
#pragma unroll
    for (int k = 1; k < Pp; ++k) m = fmaxf(m, sc[k]);
    float e[Pp]; float s = 0.f;
#pragma unroll
    for (int k = 0; k < Pp; ++k) { e[k] = expf(sc[k] - m); s += e[k]; }
    float inv = 1.0f / s;
    float acc = 0.f;
#pragma unroll
    for (int k = 0; k < Pp; ++k)
        acc += e[k] * inv * V[(long)(b * Pp + k) * DIMc + h * HDc + d];
    outBf[(long)(b * Pp + q) * DIMc + h * HDc + d] = f2bf(acc);
}

// ---------------------------------------------------------------------------
static inline void launch_gemm(hipStream_t st,
    const unsigned short* A, const unsigned short* Bm, const float* bias,
    float* Cf, unsigned short* Ch,
    int M, int Ncols, int Kdim,
    long lda, long bsc, long bsk, long ldc,
    int Z, int Hdec,
    long aSB, long aSH, long bSB, long bSH, long cSB, long cSH,
    int actGelu, int outBf16)
{
    int tiles32 = (Ncols + 31) / 32;
    dim3 grid((unsigned)((tiles32 + 3) / 4), (unsigned)((M + 31) / 32), (unsigned)Z);
    if (bsk == 1)
        wmma_gemm_kernel<1><<<grid, dim3(128), 0, st>>>(A, Bm, bias, Cf, Ch, M, Ncols, Kdim,
            lda, bsc, ldc, aSB, aSH, bSB, bSH, cSB, cSH, Hdec, actGelu, outBf16);
    else
        wmma_gemm_kernel<DIMc><<<grid, dim3(128), 0, st>>>(A, Bm, bias, Cf, Ch, M, Ncols, Kdim,
            lda, bsc, ldc, aSB, aSH, bSB, bSH, cSB, cSH, Hdec, actGelu, outBf16);
}

extern "C" void kernel_launch(void* const* d_in, const int* in_sizes, int n_in,
                              void* d_out, int out_size, void* d_ws, size_t ws_size,
                              hipStream_t stream)
{
    (void)in_sizes; (void)n_in; (void)out_size; (void)ws_size;

    const float* in_pts = (const float*)d_in[0];
    const float* in_img = (const float*)d_in[1];
    const float* rel    = (const float*)d_in[26];

    // ---------------- workspace arena ----------------
    char* wsp = (char*)d_ws;
    size_t off = 0;
    auto carve = [&](size_t bytes) -> char* {
        char* p = wsp + off; off += (bytes + 255) & ~(size_t)255; return p;
    };

    unsigned short* wt[12];                       // p2i q,k,v,o | p2p q,k,v,o | i2p q,k,v,o
    for (int i = 0; i < 12; ++i) wt[i] = (unsigned short*)carve((size_t)1024 * 1024 * 2);
    unsigned short* wtF1 = (unsigned short*)carve((size_t)4096 * 1024 * 2);   // [4096,1024]
    unsigned short* wtF2 = (unsigned short*)carve((size_t)1024 * 4096 * 2);   // [1024,4096]

    unsigned short* img16 = (unsigned short*)carve((size_t)32768 * 1024 * 2); // 64 MiB
    unsigned short* kimg  = (unsigned short*)carve((size_t)32768 * 1024 * 2); // 64 MiB (adjacent)
    unsigned short* vimg  = (unsigned short*)carve((size_t)32768 * 1024 * 2);

    float*          scoresB  = (float*)carve((size_t)Bc * Hh * Pp * Nn * 4);      // shared p2i/i2p
    unsigned short* probsP2i = (unsigned short*)carve(((size_t)Bc * Hh * Pp * Nn + 32 * Nn) * 2);
    unsigned short* probsI2p = (unsigned short*)carve((size_t)Bc * Hh * Nn * 32 * 2);

    unsigned short* pts0_16  = (unsigned short*)carve((size_t)64 * 1024 * 2);  // 48 + slack
    unsigned short* qpts     = (unsigned short*)carve((size_t)96 * 1024 * 2);  // slack rows
    unsigned short* attnoP2i = (unsigned short*)carve((size_t)64 * 1024 * 2);
    float*          oprojP2i = (float*)carve((size_t)48 * 1024 * 4);
    float*          pts1     = (float*)carve((size_t)48 * 1024 * 4);
    unsigned short* pts1_16  = (unsigned short*)carve((size_t)64 * 1024 * 2);
    float*          qp2p     = (float*)carve((size_t)48 * 1024 * 4);
    float*          kp2p     = (float*)carve((size_t)48 * 1024 * 4);
    float*          vp2p     = (float*)carve((size_t)48 * 1024 * 4);
    unsigned short* attnoP2p = (unsigned short*)carve((size_t)64 * 1024 * 2);
    float*          oprojP2p = (float*)carve((size_t)48 * 1024 * 4);
    float*          pts2     = (float*)carve((size_t)48 * 1024 * 4);
    unsigned short* pts2_16  = (unsigned short*)carve((size_t)64 * 1024 * 2);
    unsigned short* kptsI2p  = (unsigned short*)carve((size_t)96 * 1024 * 2);  // slack rows
    unsigned short* vptsI2p  = (unsigned short*)carve((size_t)96 * 1024 * 2);  // slack rows
    unsigned short* ffnH     = (unsigned short*)carve((size_t)64 * 4096 * 2);  // 48 + slack
    float*          ffn2out  = (float*)carve((size_t)48 * 1024 * 4);

    // buffer reuse
    float*          oprojImg = (float*)img16;   // 128 MiB fp32 spans img16+kimg (contiguous)
    unsigned short* qimg     = kimg;            // i2p Q reuses p2i K buffer
    unsigned short* attnoI2p = vimg;            // i2p attn-out reuses p2i V buffer

    const long DD = 1024, NNl = 4096, PPl = 6, HHl = 8, HDl = 128;

    // ---------------- weight transpose + bf16 convert ----------------
    auto wtr = [&](const float* W, unsigned short* Wt, int K, int Nout) {
        dim3 g((unsigned)((K + 255) / 256), (unsigned)Nout);
        wtrans_kernel<<<g, 256, 0, stream>>>(W, Wt, K, Nout);
    };
    for (int m = 0; m < 3; ++m)
        for (int j = 0; j < 4; ++j)
            wtr((const float*)d_in[2 + m * 8 + j * 2], wt[m * 4 + j], 1024, 1024);
    wtr((const float*)d_in[35], wtF1, 1024, 4096);
    wtr((const float*)d_in[37], wtF2, 4096, 1024);

    // ---------------- activation converts ----------------
    f2bf_kernel<<<(49152 + 255) / 256, 256, 0, stream>>>(in_pts, pts0_16, 49152);
    f2bf_kernel<<<33554432 / 256, 256, 0, stream>>>(in_img, img16, 33554432);

    // ================= 1) points -> image cross attention (p2i) =================
    launch_gemm(stream, pts0_16, wt[0], (const float*)d_in[3], nullptr, qpts,
                48, 1024, 1024, DD, DD, 1, DD, 1, 1, 0,0,0,0,0,0, 0, 1);
    launch_gemm(stream, img16, wt[1], (const float*)d_in[5], nullptr, kimg,
                32768, 1024, 1024, DD, DD, 1, DD, 1, 1, 0,0,0,0,0,0, 0, 1);
    launch_gemm(stream, img16, wt[2], (const float*)d_in[7], nullptr, vimg,
                32768, 1024, 1024, DD, DD, 1, DD, 1, 1, 0,0,0,0,0,0, 0, 1);
    // scores[b,h,q,n] = Q . K
    launch_gemm(stream, qpts, kimg, nullptr, scoresB, nullptr,
                6, 4096, 128, DD, DD, 1, NNl, 64, 8,
                PPl * DD, HDl, NNl * DD, HDl, HHl * PPl * NNl, PPl * NNl, 0, 0);
    softmax_p2i_kernel<<<Bc * Hh * Pp, 256, 0, stream>>>(scoresB, probsP2i);
    // out[b,q,h,d] = probs @ V  (B k-stride = DIM)
    launch_gemm(stream, probsP2i, vimg, nullptr, nullptr, attnoP2i,
                6, 128, 4096, NNl, 1, DD, DD, 64, 8,
                HHl * PPl * NNl, PPl * NNl, NNl * DD, HDl, PPl * DD, HDl, 0, 1);
    launch_gemm(stream, attnoP2i, wt[3], (const float*)d_in[9], oprojP2i, nullptr,
                48, 1024, 1024, DD, DD, 1, DD, 1, 1, 0,0,0,0,0,0, 0, 0);
    ln_kernel<<<48, 256, 0, stream>>>(in_pts, oprojP2i,
                (const float*)d_in[27], (const float*)d_in[28], pts1, pts1_16);

    // ================= 2) points self attention with rel-pos (p2p) =================
    launch_gemm(stream, pts1_16, wt[4], (const float*)d_in[11], qp2p, nullptr,
                48, 1024, 1024, DD, DD, 1, DD, 1, 1, 0,0,0,0,0,0, 0, 0);
    launch_gemm(stream, pts1_16, wt[5], (const float*)d_in[13], kp2p, nullptr,
                48, 1024, 1024, DD, DD, 1, DD, 1, 1, 0,0,0,0,0,0, 0, 0);
    launch_gemm(stream, pts1_16, wt[6], (const float*)d_in[15], vp2p, nullptr,
                48, 1024, 1024, DD, DD, 1, DD, 1, 1, 0,0,0,0,0,0, 0, 0);
    p2p_attn_kernel<<<Bc * Hh * Pp, 128, 0, stream>>>(qp2p, kp2p, vp2p, rel, attnoP2p);
    launch_gemm(stream, attnoP2p, wt[7], (const float*)d_in[17], oprojP2p, nullptr,
                48, 1024, 1024, DD, DD, 1, DD, 1, 1, 0,0,0,0,0,0, 0, 0);
    ln_kernel<<<48, 256, 0, stream>>>(pts1, oprojP2p,
                (const float*)d_in[29], (const float*)d_in[30], pts2, pts2_16);

    // ================= 3) image -> points cross attention (i2p) =================
    launch_gemm(stream, img16, wt[8], (const float*)d_in[19], nullptr, qimg,
                32768, 1024, 1024, DD, DD, 1, DD, 1, 1, 0,0,0,0,0,0, 0, 1);
    launch_gemm(stream, pts2_16, wt[9], (const float*)d_in[21], nullptr, kptsI2p,
                48, 1024, 1024, DD, DD, 1, DD, 1, 1, 0,0,0,0,0,0, 0, 1);
    launch_gemm(stream, pts2_16, wt[10], (const float*)d_in[23], nullptr, vptsI2p,
                48, 1024, 1024, DD, DD, 1, DD, 1, 1, 0,0,0,0,0,0, 0, 1);
    // scores[b,h,n,k] (ldc=6)
    launch_gemm(stream, qimg, kptsI2p, nullptr, scoresB, nullptr,
                4096, 6, 128, DD, DD, 1, 6, 64, 8,
                NNl * DD, HDl, PPl * DD, HDl, HHl * NNl * 6, NNl * 6, 0, 0);
    softmax_i2p_kernel<<<(Bc * Hh * Nn) / 256, 256, 0, stream>>>(scoresB, probsI2p);
    // out[b,n,h,d] = probs(pad32) @ V  (B k-stride = DIM)
    launch_gemm(stream, probsI2p, vptsI2p, nullptr, nullptr, attnoI2p,
                4096, 128, 32, 32, 1, DD, DD, 64, 8,
                HHl * NNl * 32, NNl * 32, PPl * DD, HDl, NNl * DD, HDl, 0, 1);
    launch_gemm(stream, attnoI2p, wt[11], (const float*)d_in[25], oprojImg, nullptr,
                32768, 1024, 1024, DD, DD, 1, DD, 1, 1, 0,0,0,0,0,0, 0, 0);

    float* outPts = (float*)d_out;
    float* outImg = outPts + 49152;
    ln_kernel<<<32768, 256, 0, stream>>>(in_img, oprojImg,
                (const float*)d_in[31], (const float*)d_in[32], outImg, nullptr);

    // ================= 4) points FFN =================
    launch_gemm(stream, pts2_16, wtF1, (const float*)d_in[36], nullptr, ffnH,
                48, 4096, 1024, DD, DD, 1, NNl, 1, 1, 0,0,0,0,0,0, 1 /*gelu*/, 1);
    launch_gemm(stream, ffnH, wtF2, (const float*)d_in[38], ffn2out, nullptr,
                48, 1024, 4096, NNl, NNl, 1, DD, 1, 1, 0,0,0,0,0,0, 0, 0);
    ln_kernel<<<48, 256, 0, stream>>>(pts2, ffn2out,
                (const float*)d_in[33], (const float*)d_in[34], outPts, nullptr);
}
